// NCDEModel_20160576487720
// MI455X (gfx1250) — matrix operand order repaired and checked by reading, not compiled, not run on hardware
//
#include <hip/hip_runtime.h>

// ---------------------------------------------------------------------------
// NCDE: spline-derivative precompute + 64-step WMMA ODE scan  (gfx1250)
// ---------------------------------------------------------------------------

typedef __attribute__((ext_vector_type(16))) __bf16 v16bf;
typedef __attribute__((ext_vector_type(8)))  float  v8f;

static __device__ __forceinline__ unsigned short f2bf(float f) {
  return __builtin_bit_cast(unsigned short, (__bf16)f);
}
static __device__ __forceinline__ float fast_tanh(float x) {
  // tanh(x) = 1 - 2/(exp(2x)+1);  exp(2x) = exp2(x * 2/ln2)
  float e = __builtin_amdgcn_exp2f(x * 2.885390081777927f);
  return 1.0f - 2.0f * __builtin_amdgcn_rcpf(e + 1.0f);
}
static __device__ __forceinline__ void lds_fence() {
  asm volatile("s_wait_dscnt 0" ::: "memory");
}
static __device__ __forceinline__ void async_fence() {
  asm volatile("s_wait_asynccnt 0x0" ::: "memory");
}

// ---------------------------------------------------------------------------
// Phase 1: natural cubic spline M + derivative grid, pre-scaled by dt.
// One thread per (b, c) series.  dxs layout: [b][k][c]  (B x 64 x 8 f32)
// ---------------------------------------------------------------------------
__global__ __launch_bounds__(256) void ncde_spline_kernel(
    const float* __restrict__ X, float* __restrict__ dxs) {
  const int gid = blockIdx.x * 256 + threadIdx.x;
  const int b = gid >> 3;
  const int c = gid & 7;

  const float* xp = X + b * 512 + c;     // (b*T + t)*C + c , T=64, C=8
  float y[64];
#pragma unroll
  for (int t = 0; t < 64; ++t) y[t] = xp[t * 8];

  // Thomas solve of tri(1,4,1), unknowns M[1..62]; M[0]=M[63]=0.
  float M[64];
  float cp[63];
  M[0] = 0.0f; M[63] = 0.0f;
  float prevcp = 0.0f, prevdp = 0.0f;
#pragma unroll
  for (int i = 1; i <= 62; ++i) {
    float r = 6.0f * (y[i - 1] + y[i + 1] - 2.0f * y[i]);
    float w = 1.0f / (4.0f - prevcp);      // pivots are data-independent -> folds
    prevcp = w;
    cp[i] = w;
    prevdp = (r - prevdp) * w;
    M[i] = prevdp;
  }
#pragma unroll
  for (int i = 61; i >= 1; --i) M[i] = M[i] - cp[i] * M[i + 1];

  const float dt = 63.0f / 64.0f;
  float* op = dxs + b * 512 + c;
#pragma unroll
  for (int k = 0; k < 64; ++k) {
    int num = k * 63;
    int idx = num >> 6;                    // floor(k*63/64), max 62
    float u = (float)(num - (idx << 6)) * (1.0f / 64.0f);
    float om = 1.0f - u;
    float d = -M[idx] * om * om * 0.5f + M[idx + 1] * u * u * 0.5f
              + (y[idx + 1] - y[idx]) - (M[idx + 1] - M[idx]) * (1.0f / 6.0f);
    op[k * 8] = d * dt;
  }
}

// ---------------------------------------------------------------------------
// Phase 2: h0 = X[:,0,:]@Wp + bp, then 64-step scan with bf16 WMMA GEMMs.
// 256 threads = 8 waves; each wave owns 16 rows of B; weights live in LDS.
// W2 is stored output-permuted (N' = c*64 + hh) so the F·dX contraction is
// lane-local against the C-layout of h.  The per-step dX tile is fetched
// with GLOBAL_LOAD_ASYNC_TO_LDS (ASYNCcnt) and overlapped with GEMM1+tanh.
// ---------------------------------------------------------------------------
__global__ __launch_bounds__(256) void ncde_scan_kernel(
    const float* __restrict__ X,  const float* __restrict__ Wp,
    const float* __restrict__ bp, const float* __restrict__ W1,
    const float* __restrict__ b1, const float* __restrict__ W2,
    const float* __restrict__ b2, const float* __restrict__ dxs,
    float* __restrict__ out) {
  // --- LDS: pre-swizzled bf16 weight fragments + per-wave staging ----------
  __shared__ __align__(32) unsigned short w1s[16 * 32 * 16];    //  16 KB
  __shared__ __align__(32) unsigned short w2s[128 * 32 * 16];   // 128 KB
  __shared__ __align__(32) unsigned short astg[8][4 * 32 * 16]; //  32 KB
  __shared__ float dbuf[8][128];                                //   4 KB

  const int tid  = threadIdx.x;
  const int lane = tid & 31;
  const int wv   = tid >> 5;
  const int hw   = lane >> 4;    // half-wave
  const int n    = lane & 15;    // N index within tile

  // ---- fill W1 fragments: tile t1 = kk*8 + nt ; b[i]: K=kk*32+hw*8+(i&7)+(i&8)*2 -----
  for (int idx = tid; idx < 16 * 512; idx += 256) {
    int i = idx & 15, l = (idx >> 4) & 31, t = idx >> 9;
    int kk = t >> 3, nt = t & 7;
    int K = kk * 32 + (l >> 4) * 8 + (i & 7) + ((i & 8) << 1);
    int N = nt * 16 + (l & 15);
    w1s[idx] = f2bf(W1[K * 128 + N]);
  }
  // ---- fill W2 fragments, output permuted: N' = c*64 + hh -----------------
  for (int idx = tid; idx < 128 * 512; idx += 256) {
    int i = idx & 15, l = (idx >> 4) & 31, t = idx >> 9;
    int kk = t >> 5, nt = t & 31;
    int K  = kk * 32 + (l >> 4) * 8 + (i & 7) + ((i & 8) << 1);
    int Np = nt * 16 + (l & 15);
    int cc = Np >> 6, hh = Np & 63;
    w2s[idx] = f2bf(W2[K * 512 + hh * 8 + cc]);
  }
  __syncthreads();

  const int b0 = (blockIdx.x * 8 + wv) * 16;

  // ---- h0 = X[:,0,:] @ Wp + bp, directly in WMMA C-layout -----------------
  float h[4][8];
#pragma unroll
  for (int j = 0; j < 4; ++j)
#pragma unroll
    for (int r = 0; r < 8; ++r) {
      float acc = bp[j * 16 + n];
      const float* xr = X + (b0 + hw * 8 + r) * 512;  // t = 0
#pragma unroll
      for (int cc = 0; cc < 8; ++cc) acc += xr[cc] * Wp[cc * 64 + j * 16 + n];
      h[j][r] = acc;
    }

  float b1reg[8];
#pragma unroll
  for (int t = 0; t < 8; ++t) b1reg[t] = b1[t * 16 + n];
  float b2p[32];                           // b2 permuted to N' = c*64 + hh
#pragma unroll
  for (int t = 0; t < 32; ++t) {
    int Np = t * 16 + n;
    b2p[t] = b2[(Np & 63) * 8 + (Np >> 6)];
  }

  unsigned short* as = astg[wv];

  for (int k = 0; k < 64; ++k) {
    // ---- async-stage this step's dX tile (16 rows x 8 ch, dt-scaled) -----
    // Lands in LDS while GEMM1 + tanh execute; waited before contraction.
#pragma unroll
    for (int q = 0; q < 4; ++q) {
      int idx = lane + 32 * q;
      const float* gp = dxs + (size_t)(b0 + (idx >> 3)) * 512 + k * 8 + (idx & 7);
      unsigned ldsa = (unsigned)(uintptr_t)&dbuf[wv][idx];
      asm volatile("global_load_async_to_lds_b32 %0, %1, off"
                   :: "v"(ldsa), "v"(gp) : "memory");
    }

    // ---- stage h as bf16 A-operand (ISA 16-bit A 16x32 layout) -----------
#pragma unroll
    for (int j = 0; j < 4; ++j)
#pragma unroll
      for (int r = 0; r < 8; ++r) {
        int col = j * 16 + n;                 // K index (H dim)
        int kk = col >> 5, c32 = col & 31;
        int hwp = (c32 >> 3) & 1;
        int i = (c32 & 7) + ((c32 & 16) >> 1);
        int lanep = hwp * 16 + hw * 8 + r;    // dest lane holds row M
        as[(kk * 32 + lanep) * 16 + i] = f2bf(h[j][r]);
      }
    lds_fence();

    // ---- GEMM1 tile -> +b1 -> tanh -> restage as GEMM2 A operand ---------
    // (fused per 16x16 tile to keep live state at one v8f accumulator)
    v16bf a0 = *(const v16bf*)&as[(0 * 32 + lane) * 16];
    v16bf a1 = *(const v16bf*)&as[(1 * 32 + lane) * 16];
#pragma unroll
    for (int nt = 0; nt < 8; ++nt) {
      v16bf q0 = *(const v16bf*)&w1s[((0 * 8 + nt) * 32 + lane) * 16];
      v16bf q1 = *(const v16bf*)&w1s[((1 * 8 + nt) * 32 + lane) * 16];
      v8f acc = {};
      acc = __builtin_amdgcn_wmma_f32_16x16x32_bf16(false, a0, false, q0,
                                                    (short)0, acc, false, false);
      acc = __builtin_amdgcn_wmma_f32_16x16x32_bf16(false, a1, false, q1,
                                                    (short)0, acc, false, false);
#pragma unroll
      for (int r = 0; r < 8; ++r) {
        float v = fast_tanh(acc[r] + b1reg[nt]);
        int col = nt * 16 + n;                // K index (VF dim)
        int kk = col >> 5, c32 = col & 31;
        int hwp = (c32 >> 3) & 1;
        int i = (c32 & 7) + ((c32 & 16) >> 1);
        int lanep = hwp * 16 + hw * 8 + r;
        as[(kk * 32 + lanep) * 16 + i] = f2bf(v);
      }
    }
    lds_fence();

    v16bf a2[4];
#pragma unroll
    for (int kk = 0; kk < 4; ++kk)
      a2[kk] = *(const v16bf*)&as[(kk * 32 + lane) * 16];

    async_fence();   // dX tile is now resident in dbuf

    // ---- GEMM2 (c-major) + lane-local contraction into h -----------------
#pragma unroll
    for (int cc = 0; cc < 8; ++cc) {
      float d[8];
#pragma unroll
      for (int r = 0; r < 8; ++r) d[r] = dbuf[wv][(hw * 8 + r) * 8 + cc];
#pragma unroll
      for (int j = 0; j < 4; ++j) {
        int t2 = cc * 4 + j;                  // output tile in permuted N'
        v8f g = {};
#pragma unroll
        for (int kk = 0; kk < 4; ++kk) {
          v16bf bq = *(const v16bf*)&w2s[((kk * 32 + t2) * 32 + lane) * 16];
          g = __builtin_amdgcn_wmma_f32_16x16x32_bf16(false, a2[kk], false, bq,
                                                      (short)0, g, false, false);
        }
        float bb = b2p[t2];
#pragma unroll
        for (int r = 0; r < 8; ++r)
          h[j][r] += (g[r] + bb) * d[r];      // += F[b][hh*8+c] * dX[b][c] * dt
      }
    }
  }

  // ---- write hT (B x 64) ---------------------------------------------------
#pragma unroll
  for (int j = 0; j < 4; ++j)
#pragma unroll
    for (int r = 0; r < 8; ++r)
      out[(b0 + hw * 8 + r) * 64 + j * 16 + n] = h[j][r];
}

// ---------------------------------------------------------------------------
extern "C" void kernel_launch(void* const* d_in, const int* in_sizes, int n_in,
                              void* d_out, int out_size, void* d_ws, size_t ws_size,
                              hipStream_t stream) {
  const float* X  = (const float*)d_in[0];
  const float* Wp = (const float*)d_in[1];
  const float* bp = (const float*)d_in[2];
  const float* W1 = (const float*)d_in[3];
  const float* b1 = (const float*)d_in[4];
  const float* W2 = (const float*)d_in[5];
  const float* b2 = (const float*)d_in[6];
  float* out = (float*)d_out;
  float* dxs = (float*)d_ws;               // 16384*64*8 f32 = 32 MB scratch

  // 16384*8 series / 256 threads
  ncde_spline_kernel<<<512, 256, 0, stream>>>(X, dxs);
  // 16384 rows / (8 waves * 16 rows)
  ncde_scan_kernel<<<128, 256, 0, stream>>>(X, Wp, bp, W1, b1, W2, b2, dxs, out);
}
